// SequenceBlock_40501541601289
// MI455X (gfx1250) — matrix operand order repaired and checked by reading, not compiled, not run on hardware
//
#include <hip/hip_runtime.h>
#include <hip/hip_bf16.h>
#include <math.h>

// Problem constants (from reference)
#define B_   8
#define H_   128
#define L_   2048
#define N_   64
#define BL_  (B_ * L_)          // 16384 positions
#define BHL_ (B_ * H_ * L_)     // 2,097,152 elements
#define EPS_ 1e-5f

typedef __bf16 bf16;
typedef __bf16 v16bf __attribute__((ext_vector_type(16)));
typedef __bf16 v8bf  __attribute__((ext_vector_type(8)));
typedef float  v8f   __attribute__((ext_vector_type(8)));
typedef unsigned int u32x4 __attribute__((ext_vector_type(4)));
typedef int          i32x4 __attribute__((ext_vector_type(4)));
typedef int          i32x8 __attribute__((ext_vector_type(8)));

#define SHUF16(lo, hi) __builtin_shufflevector((lo), (hi), 0,1,2,3,4,5,6,7,8,9,10,11,12,13,14,15)

__device__ __forceinline__ float gelu_tanh(float x) {
  float x3 = x * x * x;
  float t = tanhf(0.7978845608028654f * (x + 0.044715f * x3));
  return 0.5f * x * (1.0f + t);
}

// Low 32 bits of a generic pointer into LDS = LDS byte offset (ISA: LDS aperture
// truncates to addr[31:0]).
__device__ __forceinline__ unsigned lds_off_of(const void* p) {
  return (unsigned)(uintptr_t)p;
}

// ---------------------------------------------------------------------------
// TDM: DMA a contiguous 32KB (16384 x bf16) weight matrix global -> LDS.
// 1-D tile descriptor: data_size=2B, tensor_dim0 = tile_dim0 = 16384,
// tensor_dim0_stride = 16384, type=2 ("image"), count=1, no multicast.
__device__ __forceinline__ void tdm_load_w32k(const bf16* gsrc, unsigned lds_off) {
  unsigned long long ga = (unsigned long long)(uintptr_t)gsrc;
  u32x4 g0;
  g0[0] = 1u;                                                 // count=1, user D#
  g0[1] = lds_off;                                            // lds_addr (bytes)
  g0[2] = (unsigned)(ga & 0xFFFFFFFFu);                       // global_addr[31:0]
  g0[3] = (unsigned)((ga >> 32) & 0x01FFFFFFu) | (2u << 30);  // addr[56:32] | type=2
  i32x8 g1;
  g1[0] = (int)(1u << 16);          // workgroup_mask=0 | data_size=1 (2 bytes)
  g1[1] = (int)(0x4000u << 16);     // tensor_dim0[15:0]=16384 at bits[63:48]
  g1[2] = (int)(1u << 16);          // tensor_dim0 hi=0 | tensor_dim1=1 (bits[95:80])
  g1[3] = (int)(0x4000u << 16);     // tensor_dim1 hi=0 | tile_dim0=16384 (bits[127:112])
  g1[4] = 1;                        // tile_dim1=1, tile_dim2=0
  g1[5] = 16384;                    // tensor_dim0_stride[31:0]
  g1[6] = 0;
  g1[7] = 0;
  i32x4 gz = {0, 0, 0, 0};
#if defined(__clang_major__) && (__clang_major__ >= 23)
  i32x8 gz8 = {0, 0, 0, 0, 0, 0, 0, 0};
  __builtin_amdgcn_tensor_load_to_lds(g0, g1, gz, gz, gz8, 0);
#else
  __builtin_amdgcn_tensor_load_to_lds(g0, g1, gz, gz, 0);
#endif
}

// Async per-lane DMA: 16 bytes global -> LDS, tracked by ASYNCcnt.
__device__ __forceinline__ void async_copy_b128(unsigned lds_dst, const void* gsrc) {
  asm volatile("global_load_async_to_lds_b128 %0, %1, off"
               :: "v"(lds_dst), "v"((unsigned long long)(uintptr_t)gsrc)
               : "memory");
}
__device__ __forceinline__ void wait_async0() {
  asm volatile("s_wait_asynccnt 0x0" ::: "memory");
}

// ---------------------------------------------------------------------------
// tt[b,o] = sum_h Wt[o,h] * t[b,h] + bt[o]
__global__ void k_tt(const float* __restrict__ t, const float* __restrict__ Wt,
                     const float* __restrict__ bt, float* __restrict__ tt) {
  int idx = blockIdx.x * blockDim.x + threadIdx.x;
  if (idx >= B_ * H_) return;
  int b = idx / H_, o = idx % H_;
  const float* w  = Wt + o * H_;
  const float* tv = t  + b * H_;
  float acc = bt[o];
#pragma unroll 4
  for (int h = 0; h < H_; ++h) acc = fmaf(w[h], tv[h], acc);
  tt[idx] = acc;
}

// ---------------------------------------------------------------------------
// S4D ZOH discretization per (h,n): w = exp(dt*A), dB = (w-1)/A, e = 2*C*dB
__global__ void k_params(const float* __restrict__ log_dt,
                         const float* __restrict__ log_A_re,
                         const float* __restrict__ A_im,
                         const float* __restrict__ C_re,
                         const float* __restrict__ C_im,
                         float* __restrict__ wre, float* __restrict__ wim,
                         float* __restrict__ e0r, float* __restrict__ e0i,
                         float* __restrict__ e1r, float* __restrict__ e1i) {
  int idx = blockIdx.x * blockDim.x + threadIdx.x;
  if (idx >= H_ * N_) return;
  int h = idx / N_;
  float dt = __expf(log_dt[h]);
  float ar = -__expf(log_A_re[idx]);
  float ai = A_im[idx];
  float er = __expf(dt * ar);
  float s, c;
  __sincosf(dt * ai, &s, &c);
  float wr = er * c, wi = er * s;
  float den = ar * ar + ai * ai;
  float gr  = wr - 1.0f;
  float dbr = (gr * ar + wi * ai) / den;   // (w-1)/A  (complex divide)
  float dbi = (wi * ar - gr * ai) / den;
  float c0r = C_re[idx],           c0i = C_im[idx];
  float c1r = C_re[H_ * N_ + idx], c1i = C_im[H_ * N_ + idx];
  wre[idx] = wr;  wim[idx] = wi;
  e0r[idx] = 2.f * (c0r * dbr - c0i * dbi);
  e0i[idx] = 2.f * (c0r * dbi + c0i * dbr);
  e1r[idx] = 2.f * (c1r * dbr - c1i * dbi);
  e1i[idx] = 2.f * (c1r * dbi + c1i * dbr);
}

// ---------------------------------------------------------------------------
// Convert the three dense weights to bf16 for WMMA
__global__ void k_wcvt(const float* __restrict__ Wk, const float* __restrict__ W1,
                       const float* __restrict__ W2,
                       bf16* __restrict__ wkb, bf16* __restrict__ w1b,
                       bf16* __restrict__ w2b) {
  int i = blockIdx.x * blockDim.x + threadIdx.x;
  if (i >= H_ * H_) return;
  wkb[i] = (bf16)Wk[i];
  w1b[i] = (bf16)W1[i];
  w2b[i] = (bf16)W2[i];
}

// ---------------------------------------------------------------------------
// u = x + tt; z = LayerNorm_H(u)  — one thread per (b,l), coalesced along l
__global__ void __launch_bounds__(256) k_pre(const float* __restrict__ x,
                                             const float* __restrict__ tt,
                                             const float* __restrict__ ln_g,
                                             const float* __restrict__ ln_b,
                                             float* __restrict__ u,
                                             float* __restrict__ z) {
  __shared__ float stt[H_], sg[H_], sb[H_];
  int p = blockIdx.x * blockDim.x + threadIdx.x;   // 0..BL_-1
  int b = p / L_, l = p % L_;
  if (threadIdx.x < H_) {
    stt[threadIdx.x] = tt[b * H_ + threadIdx.x];   // whole block shares one b
    sg[threadIdx.x]  = ln_g[threadIdx.x];
    sb[threadIdx.x]  = ln_b[threadIdx.x];
  }
  __syncthreads();
  const float* xb = x + (size_t)b * H_ * L_ + l;
  float sum = 0.f, sq = 0.f;
#pragma unroll 4
  for (int h = 0; h < H_; ++h) {
    float v = xb[(size_t)h * L_] + stt[h];
    sum += v; sq = fmaf(v, v, sq);
  }
  float mean = sum * (1.0f / H_);
  float var  = sq * (1.0f / H_) - mean * mean;
  float rstd = rsqrtf(var + EPS_);
  float* ub = u + (size_t)b * H_ * L_ + l;
  float* zb = z + (size_t)b * H_ * L_ + l;
#pragma unroll 4
  for (int h = 0; h < H_; ++h) {
    float v = xb[(size_t)h * L_] + stt[h];
    ub[(size_t)h * L_] = v;
    zb[(size_t)h * L_] = (v - mean) * rstd * sg[h] + sb[h];
  }
}

// ---------------------------------------------------------------------------
// Bidirectional diagonal-SSM scan replacing the FFT long conv.
// One wave per (b,h); lane owns states n=lane and n=lane+32.
__global__ void __launch_bounds__(128) k_scan(
    const float* __restrict__ z,
    const float* __restrict__ wre, const float* __restrict__ wim,
    const float* __restrict__ e0r, const float* __restrict__ e0i,
    const float* __restrict__ e1r, const float* __restrict__ e1i,
    const float* __restrict__ D, bf16* __restrict__ G0) {
  __shared__ float ybuf[4][L_];                 // 32 KB
  int lane  = threadIdx.x & 31;
  int wslot = threadIdx.x >> 5;
  int wgid  = blockIdx.x * 4 + wslot;           // 0..1023
  int b = wgid >> 7;
  int h = wgid & 127;
  const float* zrow = z + (size_t)(b * H_ + h) * L_;
  int pi = h * N_ + lane;
  float w1r = wre[pi],      w1i = wim[pi];
  float w2r = wre[pi + 32], w2i = wim[pi + 32];
  float a0r1 = e0r[pi], a0i1 = e0i[pi], a0r2 = e0r[pi + 32], a0i2 = e0i[pi + 32];
  float a1r1 = e1r[pi], a1i1 = e1i[pi], a1r2 = e1r[pi + 32], a1i2 = e1i[pi + 32];
  float Dh = D[h];

  // ---- forward scan ----
  float f1r = 0.f, f1i = 0.f, f2r = 0.f, f2i = 0.f;
  for (int l0 = 0; l0 < L_; l0 += 32) {
    float zc = zrow[l0 + lane];
    float yk = 0.f;
#pragma unroll
    for (int j = 0; j < 32; ++j) {
      float zj = __shfl(zc, j, 32);
      float n1r = fmaf(w1r, f1r, fmaf(-w1i, f1i, zj));
      float n1i = fmaf(w1r, f1i, w1i * f1r);
      float n2r = fmaf(w2r, f2r, fmaf(-w2i, f2i, zj));
      float n2i = fmaf(w2r, f2i, w2i * f2r);
      f1r = n1r; f1i = n1i; f2r = n2r; f2i = n2i;
      float part = a0r1 * f1r - a0i1 * f1i + a0r2 * f2r - a0i2 * f2i;
#pragma unroll
      for (int off = 16; off >= 1; off >>= 1) part += __shfl_xor(part, off, 32);
      if (lane == j) yk = part;
    }
    ybuf[wslot][l0 + lane] = yk + Dh * zc;
  }

  // ---- backward scan ----
  float r1r = 0.f, r1i = 0.f, r2r = 0.f, r2i = 0.f;
  float zcarry = 0.f;                          // z[l0+32] of previous chunk (z[L]=0)
  for (int l0 = L_ - 32; l0 >= 0; l0 -= 32) {
    float zc = zrow[l0 + lane];
    float yk = 0.f;
#pragma unroll
    for (int j = 31; j >= 0; --j) {
      float zn = (j == 31) ? zcarry : __shfl(zc, j + 1, 32);
      float n1r = fmaf(w1r, r1r, fmaf(-w1i, r1i, zn));
      float n1i = fmaf(w1r, r1i, w1i * r1r);
      float n2r = fmaf(w2r, r2r, fmaf(-w2i, r2i, zn));
      float n2i = fmaf(w2r, r2i, w2i * r2r);
      r1r = n1r; r1i = n1i; r2r = n2r; r2i = n2i;
      float part = a1r1 * r1r - a1i1 * r1i + a1r2 * r2r - a1i2 * r2i;
#pragma unroll
      for (int off = 16; off >= 1; off >>= 1) part += __shfl_xor(part, off, 32);
      if (lane == j) yk = part;
    }
    ybuf[wslot][l0 + lane] += yk;
    zcarry = __shfl(zc, 0, 32);
  }

  // ---- GELU epilogue -> bf16 activation matrix (H x BL) ----
  bf16* grow = G0 + (size_t)h * BL_ + (size_t)b * L_;
  for (int l0 = 0; l0 < L_; l0 += 32)
    grow[l0 + lane] = (bf16)gelu_tanh(ybuf[wslot][l0 + lane]);
}

// ---------------------------------------------------------------------------
// WMMA GEMM core: block = 4 waves (128 threads), full M=128 strip per wave,
// 64 N-columns per block.
//  * Activation tile staged row-major via GLOBAL_LOAD_ASYNC_TO_LDS_B128
//    (ASYNCcnt) into the front half of the W region, then LDS->LDS transposed
//    into the padded GT layout that makes B fragments two ds_load_b128.
//  * Weights then DMA'd over the (dead) staging buffer by the Tensor Data
//    Mover (TENSORcnt), issued by wave 0.
struct GemmFrag {
  v8f acc[8];
};

__device__ __forceinline__ void gemm_core(const bf16* __restrict__ Wb,
                                          const bf16* __restrict__ Gin,
                                          int p0, bf16* Wl, bf16 (*GT)[H_ + 8],
                                          GemmFrag& fr) {
  bf16* Gs = Wl;                     // 16KB staging aliases front of 32KB W region
  unsigned gs_base = lds_off_of(Gs);

  // 1) async DMA: 64 x 128 bf16 tile, row-major (row=h, 64 cols of p)
  for (int it = threadIdx.x; it < H_ * 8; it += 128) {   // 1024 x b128
    int hh = it >> 3;
    int pc = (it & 7) * 8;
    const bf16* gsrc = Gin + (size_t)hh * BL_ + p0 + pc;
    if (p0 + 64 < BL_) __builtin_prefetch(gsrc + 64, 0, 1);  // global_prefetch_b8
    async_copy_b128(gs_base + (unsigned)(hh * 64 + pc) * 2, gsrc);
  }
  wait_async0();
  __syncthreads();

  // 2) LDS->LDS transpose into padded GT
  for (int it = threadIdx.x; it < H_ * 8; it += 128) {
    int hh = it >> 3;
    int pc = (it & 7) * 8;
    v8bf v = *(const v8bf*)&Gs[hh * 64 + pc];
#pragma unroll
    for (int e = 0; e < 8; ++e) GT[pc + e][hh] = v[e];
  }
  __syncthreads();

  // 3) TDM: DMA full 128x128 bf16 W over the staging region (now dead)
  if ((threadIdx.x >> 5) == 0) {
    tdm_load_w32k(Wb, lds_off_of(Wl));
    __builtin_amdgcn_s_wait_tensorcnt(0);
  }
  __syncthreads();

  // 4) WMMA main loop
  int lane = threadIdx.x & 31;
  int wv   = threadIdx.x >> 5;
  int nloc = wv * 16 + (lane & 15);
  int q0   = (lane < 16) ? 0 : 8;     // A-fragment K sub-offset per ISA layout
  int rowA = lane & 15;
  const v8f vz = {0.f, 0.f, 0.f, 0.f, 0.f, 0.f, 0.f, 0.f};
#pragma unroll
  for (int mt = 0; mt < 8; ++mt) fr.acc[mt] = vz;

#pragma unroll
  for (int kk = 0; kk < 4; ++kk) {
    int kb = kk * 32;
    int kB = kb + ((lane < 16) ? 0 : 8);       // B-fragment K sub-offset per ISA layout (matches A)
    const bf16* bp = &GT[nloc][kB];
    v8bf blo = *(const v8bf*)bp;
    v8bf bhi = *(const v8bf*)(bp + 16);
    v16bf bfrag = SHUF16(blo, bhi);
#pragma unroll
    for (int mt = 0; mt < 8; ++mt) {
      const bf16* ap = &Wl[(size_t)(mt * 16 + rowA) * H_ + kb + q0];
      v8bf alo = *(const v8bf*)ap;
      v8bf ahi = *(const v8bf*)(ap + 16);
      v16bf afrag = SHUF16(alo, ahi);
      fr.acc[mt] = __builtin_amdgcn_wmma_f32_16x16x32_bf16(
          false, afrag, false, bfrag, (short)0, fr.acc[mt], false, false);
    }
  }
}

// GEMM1: core = Wk*G + bk + u ; g = tanh(core)*sigmoid(core) -> bf16
__global__ void __launch_bounds__(128) k_gemm_g(const bf16* __restrict__ Wb,
                                                const bf16* __restrict__ Gin,
                                                const float* __restrict__ bias,
                                                const float* __restrict__ u,
                                                bf16* __restrict__ Gout) {
  __shared__ __align__(16) bf16 Wl[H_ * H_];        // 32 KB (front 16KB doubles as staging)
  __shared__ __align__(16) bf16 GT[64][H_ + 8];     // ~17 KB, rows stay 16B aligned
  int p0 = blockIdx.x * 64;
  GemmFrag fr;
  gemm_core(Wb, Gin, p0, Wl, GT, fr);

  int lane = threadIdx.x & 31;
  int wv   = threadIdx.x >> 5;
  int p    = p0 + wv * 16 + (lane & 15);
  int bidx = p >> 11;                 // / L_
  int l    = p & (L_ - 1);
  int rofs = (lane < 16) ? 0 : 8;
#pragma unroll
  for (int mt = 0; mt < 8; ++mt) {
#pragma unroll
    for (int r = 0; r < 8; ++r) {
      int o = mt * 16 + rofs + r;
      float core = fr.acc[mt][r] + bias[o] + u[(size_t)(bidx * H_ + o) * L_ + l];
      float gv = tanhf(core) * (1.0f / (1.0f + __expf(-core)));
      Gout[(size_t)o * BL_ + p] = (bf16)gv;
    }
  }
}

// GEMM2/3: out = W*g + bias (+ x for out1)
__global__ void __launch_bounds__(128) k_gemm_out(const bf16* __restrict__ Wb,
                                                  const bf16* __restrict__ Gin,
                                                  const float* __restrict__ bias,
                                                  const float* __restrict__ xadd,
                                                  float* __restrict__ out) {
  __shared__ __align__(16) bf16 Wl[H_ * H_];
  __shared__ __align__(16) bf16 GT[64][H_ + 8];
  int p0 = blockIdx.x * 64;
  GemmFrag fr;
  gemm_core(Wb, Gin, p0, Wl, GT, fr);

  int lane = threadIdx.x & 31;
  int wv   = threadIdx.x >> 5;
  int p    = p0 + wv * 16 + (lane & 15);
  int bidx = p >> 11;
  int l    = p & (L_ - 1);
  int rofs = (lane < 16) ? 0 : 8;
#pragma unroll
  for (int mt = 0; mt < 8; ++mt) {
#pragma unroll
    for (int r = 0; r < 8; ++r) {
      int o = mt * 16 + rofs + r;
      size_t oi = (size_t)(bidx * H_ + o) * L_ + l;
      float v = fr.acc[mt][r] + bias[o];
      if (xadd) v += xadd[oi];
      out[oi] = v;
    }
  }
}

// ---------------------------------------------------------------------------
extern "C" void kernel_launch(void* const* d_in, const int* in_sizes, int n_in,
                              void* d_out, int out_size, void* d_ws, size_t ws_size,
                              hipStream_t stream) {
  const float* x        = (const float*)d_in[0];
  const float* t        = (const float*)d_in[1];
  const float* Wt       = (const float*)d_in[2];
  const float* bt       = (const float*)d_in[3];
  const float* ln_g     = (const float*)d_in[4];
  const float* ln_b     = (const float*)d_in[5];
  const float* log_dt   = (const float*)d_in[6];
  const float* log_A_re = (const float*)d_in[7];
  const float* A_im     = (const float*)d_in[8];
  const float* C_re     = (const float*)d_in[9];
  const float* C_im     = (const float*)d_in[10];
  const float* D        = (const float*)d_in[11];
  const float* Wk       = (const float*)d_in[12];
  const float* bk       = (const float*)d_in[13];
  const float* W1       = (const float*)d_in[14];
  const float* b1       = (const float*)d_in[15];
  const float* W2       = (const float*)d_in[16];
  const float* b2       = (const float*)d_in[17];

  float* out1 = (float*)d_out;
  float* out2 = out1 + (size_t)BHL_;

  // Carve workspace (everything we read is fully written first)
  char* wp = (char*)d_ws;
  size_t off = 0;
  auto carve = [&](size_t bytes) -> char* {
    char* r = wp + off;
    off += (bytes + 255) & ~(size_t)255;
    return r;
  };
  float* u   = (float*)carve((size_t)BHL_ * 4);
  float* z   = (float*)carve((size_t)BHL_ * 4);
  float* wre = (float*)carve((size_t)H_ * N_ * 4);
  float* wim = (float*)carve((size_t)H_ * N_ * 4);
  float* e0r = (float*)carve((size_t)H_ * N_ * 4);
  float* e0i = (float*)carve((size_t)H_ * N_ * 4);
  float* e1r = (float*)carve((size_t)H_ * N_ * 4);
  float* e1i = (float*)carve((size_t)H_ * N_ * 4);
  float* tt  = (float*)carve((size_t)B_ * H_ * 4);
  bf16* wkb  = (bf16*)carve((size_t)H_ * H_ * 2);
  bf16* w1b  = (bf16*)carve((size_t)H_ * H_ * 2);
  bf16* w2b  = (bf16*)carve((size_t)H_ * H_ * 2);
  bf16* G0   = (bf16*)carve((size_t)H_ * BL_ * 2);
  bf16* gbf  = (bf16*)carve((size_t)H_ * BL_ * 2);
  (void)ws_size; (void)in_sizes; (void)n_in; (void)out_size;

  k_tt    <<<4,   256, 0, stream>>>(t, Wt, bt, tt);
  k_params<<<32,  256, 0, stream>>>(log_dt, log_A_re, A_im, C_re, C_im,
                                    wre, wim, e0r, e0i, e1r, e1i);
  k_wcvt  <<<64,  256, 0, stream>>>(Wk, W1, W2, wkb, w1b, w2b);
  k_pre   <<<BL_ / 256, 256, 0, stream>>>(x, tt, ln_g, ln_b, u, z);
  k_scan  <<<256, 128, 0, stream>>>(z, wre, wim, e0r, e0i, e1r, e1i, D, G0);
  k_gemm_g  <<<BL_ / 64, 128, 0, stream>>>(wkb, G0, bk, u, gbf);
  k_gemm_out<<<BL_ / 64, 128, 0, stream>>>(w1b, gbf, b1, x, out1);
  k_gemm_out<<<BL_ / 64, 128, 0, stream>>>(w2b, gbf, b2, nullptr, out2);
}
